// MLPFusionLoRA_24206435680544
// MI455X (gfx1250) — matrix operand order repaired and checked by reading, not compiled, not run on hardware
//
#include <hip/hip_runtime.h>
#include <hip/hip_bf16.h>
#include <math.h>

// Problem constants (reference: M,B,N,C,H,R = 4,8,1024,768,3072,16)
#define MM 4
#define BB 8
#define NN 1024
#define CC 768
#define HH 3072
#define RR 16
#define EE 4
#define NROWS (MM * BB * NN) // 32768 token rows

typedef __attribute__((ext_vector_type(16))) __bf16 v16bf;
typedef __attribute__((ext_vector_type(8)))  float  v8f;
typedef __attribute__((ext_vector_type(4)))  int    v4i_t;
typedef __attribute__((address_space(1))) v4i_t* g_v4i_p; // global int4*
typedef __attribute__((address_space(3))) v4i_t* l_v4i_p; // LDS int4*

union FragU { uint4 u[2]; v16bf v; };

// ---- gfx1250 async global->LDS copy (ASYNCcnt path) with safe fallback ------
#if defined(__has_builtin)
# if __has_builtin(__builtin_amdgcn_global_load_async_to_lds_b128)
#  define ASYNC_G2L 1
# endif
#endif
#ifndef ASYNC_G2L
# define ASYNC_G2L 0
#endif

#if ASYNC_G2L
# if defined(__has_builtin) && __has_builtin(__builtin_amdgcn_s_wait_asynccnt)
#  define WAIT_ASYNC() __builtin_amdgcn_s_wait_asynccnt(0)
# else
#  define WAIT_ASYNC() asm volatile("s_wait_asynccnt 0x0" ::: "memory")
# endif
#else
# define WAIT_ASYNC()
#endif

__device__ __forceinline__ void copy16_g2l(__bf16* l, const __bf16* g) {
#if ASYNC_G2L
  __builtin_amdgcn_global_load_async_to_lds_b128(
      (g_v4i_p)(void*)const_cast<__bf16*>(g),
      (l_v4i_p)(void*)l, 0, 0);
#else
  *(uint4*)l = *(const uint4*)g;
#endif
}

__device__ __forceinline__ __bf16 f2bf(float f) {
  unsigned u = __float_as_uint(f);
  u += 0x7FFFu + ((u >> 16) & 1u);           // round-to-nearest-even
  unsigned short s = (unsigned short)(u >> 16);
  union { unsigned short s; __bf16 b; } cv; cv.s = s; return cv.b;
}
__device__ __forceinline__ float bf2f(__bf16 b) {
  union { __bf16 b; unsigned short s; } cv; cv.b = b;
  return __uint_as_float(((unsigned)cv.s) << 16);
}
// 32-byte fragment load (two b128s)
__device__ __forceinline__ v16bf load_frag(const __bf16* p) {
  FragU f; const uint4* q = reinterpret_cast<const uint4*>(p);
  f.u[0] = q[0]; f.u[1] = q[1]; return f.v;
}
// ISA 16-bit A-matrix 16x32 layout: lanes 0-15 hold K 0-7 / 16-23,
// lanes 16-31 hold K 8-15 / 24-31 (element i of the v16bf fragment).
__device__ __forceinline__ int a_kmap(int half, int i) {
  return (i < 8) ? (half * 8 + i) : (16 + half * 8 + (i - 8));
}

// ---------------- weight packing: fp32 row-major [n][k] -> bf16 B-fragments --
// Layout: [ct][kc][lane][i] (ct=16-col tile, kc=32-K chunk); per lane 32B chunk.
__global__ __launch_bounds__(256) void pack_b_nk_k(const float* __restrict__ src,
                                                   __bf16* __restrict__ dst,
                                                   int Nc, int Kt) {
  const int KC = Kt >> 5;
  const long long total = (long long)Nc * Kt;
  for (long long idx = (long long)blockIdx.x * 256 + threadIdx.x; idx < total;
       idx += (long long)gridDim.x * 256) {
    int i    = (int)(idx & 15);
    int lane = (int)((idx >> 4) & 31);
    long long rest = idx >> 9;
    int kc = (int)(rest % KC);
    int ct = (int)(rest / KC);
    int n = ct * 16 + (lane & 15);
    int k = kc * 32 + (lane >> 4) * 8 + (i & 7) + (i & 8) * 2;
    dst[idx] = f2bf(src[(size_t)n * Kt + k]);
  }
}
// b1_w [E,Nc,R] / b2_w [E,Nc,R] -> B [K=E*R=64][Nc] fragments
__global__ __launch_bounds__(256) void pack_b_cat_k(const float* __restrict__ src,
                                                    __bf16* __restrict__ dst,
                                                    int Nc) {
  const long long total = (long long)Nc * 64;
  for (long long idx = (long long)blockIdx.x * 256 + threadIdx.x; idx < total;
       idx += (long long)gridDim.x * 256) {
    int i    = (int)(idx & 15);
    int lane = (int)((idx >> 4) & 31);
    long long rest = idx >> 9;
    int kc = (int)(rest % 2);
    int ct = (int)(rest / 2);
    int n = ct * 16 + (lane & 15);
    int k = kc * 32 + (lane >> 4) * 8 + (i & 7) + (i & 8) * 2;   // k = e*16 + r
    int e = k >> 4, r = k & 15;
    dst[idx] = f2bf(src[((size_t)e * Nc + n) * RR + r]);
  }
}

// ---------------- router: logits->softmax->mask renorm, t1 = x.a1^T, x->bf16 --
__global__ __launch_bounds__(256) void router_t1_k(
    const float* __restrict__ x, const int* __restrict__ mmask,
    const float* __restrict__ gate_w, const float* __restrict__ gate_b,
    const float* __restrict__ a1_w,
    __bf16* __restrict__ XB, float* __restrict__ T1, float* __restrict__ WF) {
  const int row = blockIdx.x;              // (m*B + b)*N + n
  const int m = row / (BB * NN);
  const int b = (row / NN) % BB;
  const int tid = threadIdx.x;
  float p[20];
#pragma unroll
  for (int j = 0; j < 20; ++j) p[j] = 0.f;
  for (int c = tid; c < CC; c += 256) {
    float xv = x[(size_t)row * CC + c];
    XB[(size_t)row * CC + c] = f2bf(xv);
#pragma unroll
    for (int r = 0; r < 16; ++r) p[r] += xv * a1_w[((size_t)m * RR + r) * CC + c];
#pragma unroll
    for (int e = 0; e < 4; ++e) p[16 + e] += xv * gate_w[((size_t)m * EE + e) * CC + c];
  }
#pragma unroll
  for (int off = 16; off > 0; off >>= 1)
#pragma unroll
    for (int j = 0; j < 20; ++j) p[j] += __shfl_xor(p[j], off, 32);
  __shared__ float red[8 * 20];
  __shared__ float tot[20];
  int wave = tid >> 5, lane = tid & 31;
  if (lane == 0) {
#pragma unroll
    for (int j = 0; j < 20; ++j) red[wave * 20 + j] = p[j];
  }
  __syncthreads();
  if (tid < 20) {
    float s = 0.f;
    for (int w = 0; w < 8; ++w) s += red[w * 20 + tid];
    tot[tid] = s;
  }
  __syncthreads();
  if (tid < 16) T1[(size_t)row * 16 + tid] = tot[tid];
  if (tid == 0) {
    float lg[4], mx = -1e30f;
#pragma unroll
    for (int e = 0; e < 4; ++e) { lg[e] = tot[16 + e] + gate_b[m * EE + e]; mx = fmaxf(mx, lg[e]); }
    float Z = 0.f, ex[4];
#pragma unroll
    for (int e = 0; e < 4; ++e) { ex[e] = expf(lg[e] - mx); Z += ex[e]; }
    float w[4], sw = 0.f;
#pragma unroll
    for (int e = 0; e < 4; ++e) { w[e] = (ex[e] / Z) * (float)mmask[e * BB + b]; sw += w[e]; }
    float self = (float)mmask[m * BB + b];
    float inv = self / (sw + 1e-6f);
#pragma unroll
    for (int e = 0; e < 4; ++e) WF[(size_t)row * 4 + e] = w[e] * inv;
  }
}

// ---------------- q[i,b,n,e*R+r] = wfull[i,b,n,e] * t[e,b,n,r]  (bf16, A-major)
__global__ __launch_bounds__(256) void make_q_k(const float* __restrict__ WF,
                                                const float* __restrict__ T,
                                                __bf16* __restrict__ Q) {
  long long idx = (long long)blockIdx.x * 256 + threadIdx.x;
  if (idx >= (long long)NROWS * 64) return;
  int er = (int)(idx & 63);
  long long row = idx >> 6;
  int e = er >> 4, r = er & 15;
  int b = (int)((row / NN) % BB);
  int n = (int)(row % NN);
  long long trow = ((long long)e * BB + b) * NN + n;
  Q[idx] = f2bf(WF[row * 4 + e] * T[trow * 16 + r]);
}

// ---- fc1 GEMM: h = gelu(x@fc1^T + q1@b1cat + b); also t2 = h@a2^T via WMMA --
__global__ __launch_bounds__(256) void gemm_fc1_k(
    const __bf16* __restrict__ XB, const __bf16* __restrict__ BP1,
    const __bf16* __restrict__ Q1, const __bf16* __restrict__ BC1,
    const float* __restrict__ fc1_b, const __bf16* __restrict__ PA2,
    __bf16* __restrict__ HB, float* __restrict__ T2) {
  __shared__ __align__(16) __bf16 ldsA[24 * 512]; // 16x768 A, fragment-permuted
  __shared__ __align__(16) __bf16 ldsQ[2 * 512];  // 16x64 q1
  __shared__ __align__(16) __bf16 ldsH[8 * 512];  // per-wave 16x32 h A-fragments
  __shared__ float ldsT2[8 * 256];                // per-wave t2 partials
  const int rowBase = blockIdx.x * 16;
  const int mIdx = rowBase / (BB * NN);
  const int tid = threadIdx.x, lane = tid & 31, wave = tid >> 5;

  // b128 fills (async->LDS when available): a 16B K-run maps to one fragment slot
  for (int f = tid; f < 16 * 96; f += 256) {          // 768 K / 8 per chunk
    int row = f / 96, kb = f % 96;
    int kc = kb >> 2, rem = kb & 3;
    int l2 = ((rem & 1) << 4) | row, ip = (rem >> 1) << 3;
    copy16_g2l(&ldsA[kc * 512 + l2 * 16 + ip], &XB[(size_t)(rowBase + row) * CC + kb * 8]);
  }
  for (int f = tid; f < 16 * 8; f += 256) {           // 64 K of q1
    int row = f / 8, kb = f % 8;
    int kc = kb >> 2, rem = kb & 3;
    int l2 = ((rem & 1) << 4) | row, ip = (rem >> 1) << 3;
    copy16_g2l(&ldsQ[kc * 512 + l2 * 16 + ip], &Q1[(size_t)(rowBase + row) * 64 + kb * 8]);
  }
  WAIT_ASYNC();
  __syncthreads();

  const __bf16* PA2m = PA2 + (size_t)mIdx * 16 * HH;  // a2 B-fragments, this modality
  const __bf16* aBase = &ldsA[lane * 16];             // A fragments: +kc*512
  const __bf16* qBase = &ldsQ[lane * 16];
  const __bf16* hBase = &ldsH[(size_t)wave * 512 + lane * 16];
  const __bf16* pBase = PA2m + (size_t)wave * 12 * 512 + lane * 16;
  v8f t2acc = {0.f, 0.f, 0.f, 0.f, 0.f, 0.f, 0.f, 0.f};
  const int mbase = (lane >> 4) * 8;                  // C/D layout: VGPR v -> M=v(+8)

  for (int ci = 0; ci < 24; ++ci) {                   // wave owns 24 contiguous tiles
    const int ct = wave * 24 + ci;
    // loop-invariant bases -> loads use immediate offsets, no per-frag VALU
    const __bf16* bBase = BP1 + (size_t)ct * (24 * 512) + lane * 16;
    __builtin_prefetch(bBase + 24 * 512, 0, 3);       // next tile's B region
    v8f acc = {0.f, 0.f, 0.f, 0.f, 0.f, 0.f, 0.f, 0.f};
#pragma unroll
    for (int kc = 0; kc < 24; ++kc) {
      v16bf a  = load_frag(aBase + kc * 512);
      v16bf bm = load_frag(bBase + kc * 512);
      acc = __builtin_amdgcn_wmma_f32_16x16x32_bf16(false, a, false, bm,
                                                    (short)0, acc, false, false);
    }
    const __bf16* cBase = BC1 + (size_t)ct * (2 * 512) + lane * 16;
#pragma unroll
    for (int kc = 0; kc < 2; ++kc) {
      v16bf a  = load_frag(qBase + kc * 512);
      v16bf bm = load_frag(cBase + kc * 512);
      acc = __builtin_amdgcn_wmma_f32_16x16x32_bf16(false, a, false, bm,
                                                    (short)0, acc, false, false);
    }
    // epilogue: bias + exact GELU; emit h to global and into A-fragment LDS
    int col = ct * 16 + (lane & 15);
    float bias = fc1_b[col];
    int colLocal = ((ci & 1) << 4) | (lane & 15);     // col within 32-wide pair
    int h2 = (colLocal >> 3) & 1;
    int ip = (colLocal & 7) + ((colLocal & 16) ? 8 : 0);
#pragma unroll
    for (int v = 0; v < 8; ++v) {
      float xv = acc[v] + bias;
      float g = 0.5f * xv * (1.0f + erff(xv * 0.70710678118f));
      __bf16 gb = f2bf(g);
      HB[(size_t)(rowBase + mbase + v) * HH + col] = gb;
      ldsH[wave * 512 + ((h2 << 4) | (mbase + v)) * 16 + ip] = gb;
    }
    if (ci & 1) {                                     // pair complete: t2 += h@a2^T
      v16bf a  = load_frag(hBase);
      v16bf bm = load_frag(pBase + (size_t)(ci >> 1) * 512);
      t2acc = __builtin_amdgcn_wmma_f32_16x16x32_bf16(false, a, false, bm,
                                                      (short)0, t2acc, false, false);
    }
  }
  // cross-wave reduce t2 (16 rows x 16 r)
#pragma unroll
  for (int v = 0; v < 8; ++v)
    ldsT2[wave * 256 + (mbase + v) * 16 + (lane & 15)] = t2acc[v];
  __syncthreads();
  {
    float s = 0.f;
    for (int w = 0; w < 8; ++w) s += ldsT2[w * 256 + tid];
    T2[(size_t)(rowBase + (tid >> 4)) * 16 + (tid & 15)] = s;
  }
}

// ---------------- fc2 GEMM: out = h@fc2^T + q2@b2cat + b ---------------------
__global__ __launch_bounds__(256) void gemm_fc2_k(
    const __bf16* __restrict__ HB, const __bf16* __restrict__ BP2,
    const __bf16* __restrict__ Q2, const __bf16* __restrict__ BC2,
    const float* __restrict__ fc2_b, float* __restrict__ out) {
  __shared__ __align__(16) __bf16 ldsA[24 * 512]; // 16x768 K-chunk of h
  __shared__ __align__(16) __bf16 ldsQ[2 * 512];
  const int rowBase = blockIdx.x * 16;
  const int tid = threadIdx.x, lane = tid & 31, wave = tid >> 5;
  for (int f = tid; f < 16 * 8; f += 256) {
    int row = f / 8, kb = f % 8;
    int kc = kb >> 2, rem = kb & 3;
    int l2 = ((rem & 1) << 4) | row, ip = (rem >> 1) << 3;
    copy16_g2l(&ldsQ[kc * 512 + l2 * 16 + ip], &Q2[(size_t)(rowBase + row) * 64 + kb * 8]);
  }
  WAIT_ASYNC();
  const __bf16* aBase = &ldsA[lane * 16];
  const __bf16* qBase = &ldsQ[lane * 16];
  v8f acc[6];
#pragma unroll
  for (int ci = 0; ci < 6; ++ci) acc[ci] = (v8f){0.f,0.f,0.f,0.f,0.f,0.f,0.f,0.f};
  for (int chunk = 0; chunk < 4; ++chunk) {        // H = 4 * 768
    __syncthreads();
    for (int f = tid; f < 16 * 96; f += 256) {
      int row = f / 96, kb = f % 96;
      int kc = kb >> 2, rem = kb & 3;
      int l2 = ((rem & 1) << 4) | row, ip = (rem >> 1) << 3;
      copy16_g2l(&ldsA[kc * 512 + l2 * 16 + ip],
                 &HB[(size_t)(rowBase + row) * HH + chunk * 768 + kb * 8]);
    }
    WAIT_ASYNC();
    __syncthreads();
    for (int ci = 0; ci < 6; ++ci) {
      const int ct = wave + ci * 8;                 // 48 col tiles over 8 waves
      const __bf16* bBase = BP2 + ((size_t)ct * 96 + chunk * 24) * 512 + lane * 16;
      __builtin_prefetch(bBase + 24 * 512, 0, 3);
#pragma unroll
      for (int kc = 0; kc < 24; ++kc) {
        v16bf a  = load_frag(aBase + kc * 512);
        v16bf bm = load_frag(bBase + kc * 512);
        acc[ci] = __builtin_amdgcn_wmma_f32_16x16x32_bf16(false, a, false, bm,
                                                          (short)0, acc[ci], false, false);
      }
    }
  }
  for (int ci = 0; ci < 6; ++ci) {
    const int ct = wave + ci * 8;
    const __bf16* cBase = BC2 + (size_t)ct * (2 * 512) + lane * 16;
#pragma unroll
    for (int kc = 0; kc < 2; ++kc) {
      v16bf a  = load_frag(qBase + kc * 512);
      v16bf bm = load_frag(cBase + kc * 512);
      acc[ci] = __builtin_amdgcn_wmma_f32_16x16x32_bf16(false, a, false, bm,
                                                        (short)0, acc[ci], false, false);
    }
    int col = ct * 16 + (lane & 15);
    float bias = fc2_b[col];
    int mbase = (lane >> 4) * 8;
#pragma unroll
    for (int v = 0; v < 8; ++v)
      out[(size_t)(rowBase + mbase + v) * CC + col] = acc[ci][v] + bias;
  }
}

// ------------------------------- launcher ------------------------------------
extern "C" void kernel_launch(void* const* d_in, const int* in_sizes, int n_in,
                              void* d_out, int out_size, void* d_ws, size_t ws_size,
                              hipStream_t stream) {
  (void)in_sizes; (void)n_in; (void)out_size; (void)ws_size;
  const float* x      = (const float*)d_in[0];
  const int*   mmask  = (const int*)  d_in[1];
  const float* fc1_w  = (const float*)d_in[2];
  const float* fc1_b  = (const float*)d_in[3];
  const float* fc2_w  = (const float*)d_in[4];
  const float* fc2_b  = (const float*)d_in[5];
  const float* gate_w = (const float*)d_in[6];
  const float* gate_b = (const float*)d_in[7];
  const float* a1_w   = (const float*)d_in[8];
  const float* b1_w   = (const float*)d_in[9];
  const float* a2_w   = (const float*)d_in[10];
  const float* b2_w   = (const float*)d_in[11];
  float* out = (float*)d_out;

  // Workspace carve-up (~276 MB total), 256B aligned.
  char* w = (char*)d_ws;
  size_t off = 0;
  auto carve = [&](size_t bytes) { char* p = w + off; off += (bytes + 255) & ~(size_t)255; return p; };
  __bf16* XB  = (__bf16*)carve((size_t)NROWS * CC * 2);      // x in bf16, A-row-major
  __bf16* HB  = (__bf16*)carve((size_t)NROWS * HH * 2);      // gelu(h) bf16
  __bf16* BP1 = (__bf16*)carve((size_t)CC * HH * 2);         // fc1 B-fragments
  __bf16* BP2 = (__bf16*)carve((size_t)HH * CC * 2);         // fc2 B-fragments
  __bf16* BC1 = (__bf16*)carve((size_t)64 * HH * 2);         // b1cat B-fragments
  __bf16* BC2 = (__bf16*)carve((size_t)64 * CC * 2);         // b2cat B-fragments
  __bf16* PA2 = (__bf16*)carve((size_t)MM * 16 * HH * 2);    // a2^T B-fragments (per m)
  __bf16* Q1  = (__bf16*)carve((size_t)NROWS * 64 * 2);
  __bf16* Q2  = (__bf16*)carve((size_t)NROWS * 64 * 2);
  float*  WF  = (float*) carve((size_t)NROWS * 4 * 4);       // masked renorm routing
  float*  T1  = (float*) carve((size_t)NROWS * 16 * 4);
  float*  T2  = (float*) carve((size_t)NROWS * 16 * 4);

  // 1) weight packing (L2-resident afterwards)
  pack_b_nk_k <<<9216, 256, 0, stream>>>(fc1_w, BP1, HH, CC);
  pack_b_nk_k <<<9216, 256, 0, stream>>>(fc2_w, BP2, CC, HH);
  pack_b_cat_k<<<768,  256, 0, stream>>>(b1_w, BC1, HH);
  pack_b_cat_k<<<192,  256, 0, stream>>>(b2_w, BC2, CC);
  for (int m = 0; m < MM; ++m)
    pack_b_nk_k<<<192, 256, 0, stream>>>(a2_w + (size_t)m * RR * HH,
                                         PA2 + (size_t)m * 16 * HH, 16, HH);
  // 2) routing weights + t1 + x->bf16
  router_t1_k<<<NROWS, 256, 0, stream>>>(x, mmask, gate_w, gate_b, a1_w, XB, T1, WF);
  // 3) q1 = wfull * t1
  make_q_k<<<(NROWS * 64) / 256, 256, 0, stream>>>(WF, T1, Q1);
  // 4) fc1 + lora1-combine + gelu (+ t2 = h@a2^T via WMMA, fused)
  gemm_fc1_k<<<NROWS / 16, 256, 0, stream>>>(XB, BP1, Q1, BC1, fc1_b, PA2, HB, T2);
  // 5) q2 = wfull * t2
  make_q_k<<<(NROWS * 64) / 256, 256, 0, stream>>>(WF, T2, Q2);
  // 6) fc2 + lora2-combine + bias -> out
  gemm_fc2_k<<<NROWS / 16, 256, 0, stream>>>(HB, BP2, Q2, BC2, fc2_b, out);
}
